// KStackModel_68487548502452
// MI455X (gfx1250) — matrix operand order repaired and checked by reading, not compiled, not run on hardware
//
#include <hip/hip_runtime.h>
#include <math.h>

// ---------------- model constants ----------------
#define VSZ 32000
#define NTOK 2048
#define DDIM 1024
#define RRK 16
#define LLAY 4
#define KKS 7
#define BB 2
#define HDIM 4096
#define TTOK (BB * NTOK)   // 4096
#define GMIN_C 0.85f
#define GMAX_C 1.0f

// ---------------- vector types ----------------
typedef float  v8f   __attribute__((ext_vector_type(8)));
typedef __bf16 v16bf __attribute__((ext_vector_type(16)));
typedef __bf16 bf16x8 __attribute__((ext_vector_type(8)));

#define TKP 40   // padded LDS row stride in bf16 (80B: conflict-free, 16B aligned)

__device__ __forceinline__ float sigmoid_f(float x) { return 1.0f / (1.0f + expf(-x)); }
__device__ __forceinline__ float gelu_f(float x) {
    float x3 = x * x * x;
    return 0.5f * x * (1.0f + tanhf(0.7978845608028654f * (x + 0.044715f * x3)));
}

// =====================================================================
// WMMA GEMM: C[M][Nmat] = act(A[M][K](f32) * Bt[Nmat][K](bf16)^T + bias) + res
// block = 256 threads (8 waves), tile 128x128, K-slab = 32
// =====================================================================
__launch_bounds__(256)
__global__ void gemm_bf16_kernel(const float* __restrict__ A,
                                 const __bf16* __restrict__ Bt,
                                 const float* __restrict__ bias,
                                 const float* __restrict__ res,
                                 float* __restrict__ C,
                                 int M, int Nmat, int K, int gelu_flag)
{
    __shared__ __bf16 As[128 * TKP];
    __shared__ __bf16 Bs[128 * TKP];

    const int tid  = threadIdx.x;
    const int lane = tid & 31;
    const int wave = tid >> 5;
    const int wm   = wave >> 2;          // 0..1  (64 rows each)
    const int wn   = wave & 3;           // 0..3  (32 cols each)
    const int gm0  = blockIdx.y * 128;
    const int gn0  = blockIdx.x * 128;

    v8f acc[4][2];
#pragma unroll
    for (int i = 0; i < 4; i++)
#pragma unroll
        for (int j = 0; j < 2; j++)
#pragma unroll
            for (int g = 0; g < 8; g++) acc[i][j][g] = 0.0f;

    const int sm = tid >> 1;             // staging row 0..127
    const int sk = (tid & 1) * 16;       // staging K offset 0 or 16

    const int halfsel  = (lane < 16) ? 0 : 1;
    const int lrow     = lane & 15;

    for (int kb = 0; kb < K; kb += 32) {
        // ---- stage A (f32 -> bf16) ----
        const float* arow = A + (size_t)(gm0 + sm) * K + (kb + sk);
        float4 f0 = ((const float4*)arow)[0];
        float4 f1 = ((const float4*)arow)[1];
        float4 f2 = ((const float4*)arow)[2];
        float4 f3 = ((const float4*)arow)[3];
        bf16x8 alo = { (__bf16)f0.x, (__bf16)f0.y, (__bf16)f0.z, (__bf16)f0.w,
                       (__bf16)f1.x, (__bf16)f1.y, (__bf16)f1.z, (__bf16)f1.w };
        bf16x8 ahi = { (__bf16)f2.x, (__bf16)f2.y, (__bf16)f2.z, (__bf16)f2.w,
                       (__bf16)f3.x, (__bf16)f3.y, (__bf16)f3.z, (__bf16)f3.w };
        __bf16* adst = &As[sm * TKP + sk];
        *(bf16x8*)(adst)     = alo;
        *(bf16x8*)(adst + 8) = ahi;

        // ---- stage B (bf16 copy) ----
        const __bf16* brow = Bt + (size_t)(gn0 + sm) * K + (kb + sk);
        bf16x8 b0 = ((const bf16x8*)brow)[0];
        bf16x8 b1 = ((const bf16x8*)brow)[1];
        __bf16* bdst = &Bs[sm * TKP + sk];
        *(bf16x8*)(bdst)     = b0;
        *(bf16x8*)(bdst + 8) = b1;

        if (kb + 32 < K) {
            __builtin_prefetch(arow + 32, 0, 1);
            __builtin_prefetch(brow + 32, 0, 1);
        }
        __syncthreads();

        // ---- A fragments: ISA 16-bit A layout (K 0-7/16-23 vs 8-15/24-31) ----
        v16bf af[4];
#pragma unroll
        for (int mt = 0; mt < 4; mt++) {
            const __bf16* rp = &As[(wm * 64 + mt * 16 + lrow) * TKP];
            bf16x8 lo = *(const bf16x8*)(rp + halfsel * 8);
            bf16x8 hi = *(const bf16x8*)(rp + 16 + halfsel * 8);
            af[mt] = __builtin_shufflevector(lo, hi, 0,1,2,3,4,5,6,7,8,9,10,11,12,13,14,15);
        }
        // ---- B fragments: ISA 16-bit B layout (K 0-7/16-23 vs 8-15/24-31) ----
        v16bf bfr[2];
#pragma unroll
        for (int nt = 0; nt < 2; nt++) {
            const __bf16* rp = &Bs[(wn * 32 + nt * 16 + lrow) * TKP];
            bf16x8 lo = *(const bf16x8*)(rp + halfsel * 8);
            bf16x8 hi = *(const bf16x8*)(rp + 16 + halfsel * 8);
            bfr[nt] = __builtin_shufflevector(lo, hi, 0,1,2,3,4,5,6,7,8,9,10,11,12,13,14,15);
        }

#pragma unroll
        for (int mt = 0; mt < 4; mt++)
#pragma unroll
            for (int nt = 0; nt < 2; nt++)
                acc[mt][nt] = __builtin_amdgcn_wmma_f32_16x16x32_bf16(
                    false, af[mt], false, bfr[nt], (short)0, acc[mt][nt], false, false);
        __syncthreads();
    }

    // ---- epilogue (C/D layout: VGPR g -> M=g (lanes<16) / M=8+g) ----
    const int cm_half = (lane < 16) ? 0 : 8;
    const int cn      = lane & 15;
#pragma unroll
    for (int mt = 0; mt < 4; mt++) {
#pragma unroll
        for (int nt = 0; nt < 2; nt++) {
            int col  = gn0 + wn * 32 + nt * 16 + cn;
            int row0 = gm0 + wm * 64 + mt * 16 + cm_half;
            float bv = bias ? bias[col] : 0.0f;
#pragma unroll
            for (int g = 0; g < 8; g++) {
                int row = row0 + g;
                float v = acc[mt][nt][g] + bv;
                if (gelu_flag) v = gelu_f(v);
                if (res) v += res[(size_t)row * Nmat + col];
                C[(size_t)row * Nmat + col] = v;
            }
        }
    }
}

// =====================================================================
// embedding gather: h[t][d] = emb[tokens[t]][d]
// =====================================================================
__global__ void gather_kernel(const int* __restrict__ tokens,
                              const float* __restrict__ emb,
                              float* __restrict__ h)
{
    size_t idx = (size_t)blockIdx.x * 256 + threadIdx.x; // T*D threads
    int t = (int)(idx >> 10);
    int d = (int)(idx & 1023);
    h[idx] = emb[(size_t)tokens[t] * DDIM + d];
}

// =====================================================================
// rmsnorm: block per token
// =====================================================================
__global__ void rmsnorm_kernel(const float* __restrict__ x,
                               const float* __restrict__ w,
                               float* __restrict__ out)
{
    const int t = blockIdx.x, tid = threadIdx.x;
    __shared__ float red[256];
    float s = 0.0f;
    for (int d = tid; d < DDIM; d += 256) { float v = x[(size_t)t * DDIM + d]; s += v * v; }
    red[tid] = s; __syncthreads();
    for (int off = 128; off > 0; off >>= 1) {
        if (tid < off) red[tid] += red[tid + off];
        __syncthreads();
    }
    float scale = rsqrtf(red[0] * (1.0f / (float)DDIM) + 1e-6f);
    for (int d = tid; d < DDIM; d += 256)
        out[(size_t)t * DDIM + d] = x[(size_t)t * DDIM + d] * scale * w[d];
}

// =====================================================================
// transpose + f32->bf16: in[Rr][Cc] -> out[Cc][Rr]
// =====================================================================
__global__ void transpose_cvt_kernel(const float* __restrict__ in,
                                     __bf16* __restrict__ out,
                                     int Rr, int Cc)
{
    __shared__ float tile[32][33];
    int tx = threadIdx.x, ty = threadIdx.y;   // 32 x 8
    int c  = blockIdx.x * 32 + tx;
#pragma unroll
    for (int i = 0; i < 4; i++)
        tile[ty + i * 8][tx] = in[(size_t)(blockIdx.y * 32 + ty + i * 8) * Cc + c];
    __syncthreads();
    int co = blockIdx.y * 32 + tx;
#pragma unroll
    for (int i = 0; i < 4; i++)
        out[(size_t)(blockIdx.x * 32 + ty + i * 8) * Rr + co] = (__bf16)tile[tx][ty + i * 8];
}

// =====================================================================
// per-layer params: alpha, gate, decay table gamma_r^n
// =====================================================================
__global__ void decay_params_kernel(const float* __restrict__ dlog,
                                    const float* __restrict__ alog,
                                    const float* __restrict__ glog,
                                    int layer,
                                    float* __restrict__ alpha,
                                    float* __restrict__ gate,
                                    float* __restrict__ decay)
{
    int idx = blockIdx.x * 256 + threadIdx.x;
    if (blockIdx.x == 0) {
        if (threadIdx.x < RRK) alpha[threadIdx.x] = sigmoid_f(alog[threadIdx.x]); // ACAP=1
        if (threadIdx.x == 0)  gate[0] = sigmoid_f(glog[layer]);
    }
    if (idx < RRK * NTOK) {
        int r = idx / NTOK, n = idx % NTOK;
        float g = GMIN_C + (GMAX_C - GMIN_C) * sigmoid_f(dlog[r]);
        decay[idx] = expf((float)n * logf(g));
    }
}

// =====================================================================
// q/k low-rank projection + l2norm (+alpha fold into q): block per token
// =====================================================================
__global__ void qk_kernel(const float* __restrict__ hn,
                          const float* __restrict__ U,
                          const float* __restrict__ Vw,
                          const float* __restrict__ alpha,
                          float* __restrict__ qout,
                          float* __restrict__ kout)
{
    const int t = blockIdx.x, tid = threadIdx.x;
    __shared__ float hrow[DDIM];
    __shared__ float part[256];
    __shared__ float vec[RRK];
    for (int d = tid; d < DDIM; d += 256) hrow[d] = hn[(size_t)t * DDIM + d];
    __syncthreads();
    const int r = tid & 15, chunk = tid >> 4;      // 16 chunks x 64 d
    for (int pass = 0; pass < 2; pass++) {
        const float* W = pass ? Vw : U;
        float s = 0.0f;
#pragma unroll 4
        for (int dd = 0; dd < 64; dd++) {
            int d = chunk * 64 + dd;
            s += hrow[d] * W[d * RRK + r];
        }
        part[tid] = s; __syncthreads();
        if (tid < RRK) {
            float a = 0.0f;
            for (int c2 = 0; c2 < 16; c2++) a += part[c2 * 16 + tid];
            vec[tid] = a;
        }
        __syncthreads();
        if (tid < RRK) {
            float n2 = 0.0f;
            for (int rr = 0; rr < RRK; rr++) { float x = vec[rr]; n2 += x * x; }
            float inv = 1.0f / fmaxf(sqrtf(n2), 1e-8f);
            if (pass == 0) qout[(size_t)t * RRK + tid] = alpha[tid] * vec[tid] * inv;
            else           kout[(size_t)t * RRK + tid] = vec[tid] * inv;
        }
        __syncthreads();
    }
}

// =====================================================================
// gated depthwise causal conv: att = gate * conv(hn)
// =====================================================================
__global__ void conv_gate_kernel(const float* __restrict__ hn,
                                 const float* __restrict__ kern,
                                 const float* __restrict__ gate,
                                 float* __restrict__ att)
{
    size_t idx = (size_t)blockIdx.x * 256 + threadIdx.x;  // T*D
    int d = (int)(idx & 1023);
    size_t tt = idx >> 10;                // global token
    int t = (int)(tt & (NTOK - 1));       // position within batch (NTOK is pow2)
    float acc = 0.0f;
#pragma unroll
    for (int k = 0; k < KKS; k++)
        if (t - k >= 0) acc += kern[k] * hn[(tt - (size_t)k) * DDIM + d];
    att[idx] = gate[0] * acc;
}

// =====================================================================
// scores[b][i][j] = (j<=i) ? sum_r qa[i][r]*gamma_r^(i-j)*k[j][r] : 0
// block = 16x16 tile of (i,j)
// =====================================================================
__global__ void scores_kernel(const float* __restrict__ q,
                              const float* __restrict__ kk,
                              const float* __restrict__ decay,
                              float* __restrict__ sc)
{
    const int b  = blockIdx.z;
    const int i0 = blockIdx.y * 16, j0 = blockIdx.x * 16;
    const int tid = threadIdx.x;
    const int ti = tid >> 4, tj = tid & 15;
    __shared__ float qs[16][16], ks[16][16];
    qs[ti][tj] = q[((size_t)(b * NTOK + i0 + ti)) * RRK + tj];
    ks[ti][tj] = kk[((size_t)(b * NTOK + j0 + ti)) * RRK + tj];
    __syncthreads();
    int i = i0 + ti, j = j0 + tj;
    float s = 0.0f;
    if (j <= i) {
        int diff = i - j;
#pragma unroll
        for (int r = 0; r < RRK; r++)
            s += qs[ti][r] * ks[tj][r] * decay[r * NTOK + diff];
    }
    sc[((size_t)b * NTOK + i) * NTOK + j] = s;
}

// =====================================================================
// host-side helpers
// =====================================================================
static void launch_gemm(const float* A, const __bf16* Bt, const float* bias,
                        const float* res, float* C, int M, int Nmat, int K,
                        int gelu, hipStream_t s)
{
    dim3 grid(Nmat / 128, M / 128);
    gemm_bf16_kernel<<<grid, 256, 0, s>>>(A, Bt, bias, res, C, M, Nmat, K, gelu);
}
static void launch_transpose(const float* in, __bf16* out, int Rr, int Cc, hipStream_t s)
{
    transpose_cvt_kernel<<<dim3(Cc / 32, Rr / 32), dim3(32, 8), 0, s>>>(in, out, Rr, Cc);
}
// k1 block: h = h + mlp(rmsnorm(h, nw))
static void run_k1(float* h, float* hn, float* mid, __bf16* wstage,
                   const float* nw, const float* w1, const float* b1,
                   const float* w2, const float* b2, hipStream_t s)
{
    rmsnorm_kernel<<<TTOK, 256, 0, s>>>(h, nw, hn);
    launch_transpose(w1, wstage, DDIM, HDIM, s);
    launch_gemm(hn, wstage, b1, nullptr, mid, TTOK, HDIM, DDIM, 1, s);
    launch_transpose(w2, wstage, HDIM, DDIM, s);
    launch_gemm(mid, wstage, b2, h, h, TTOK, DDIM, HDIM, 0, s);
}

extern "C" void kernel_launch(void* const* d_in, const int* in_sizes, int n_in,
                              void* d_out, int out_size, void* d_ws, size_t ws_size,
                              hipStream_t stream)
{
    const int*   tokens  = (const int*)  d_in[0];
    const float* emb     = (const float*)d_in[1];
    const float* k1a_nw  = (const float*)d_in[2];
    const float* k1a_w1  = (const float*)d_in[3];
    const float* k1a_b1  = (const float*)d_in[4];
    const float* k1a_w2  = (const float*)d_in[5];
    const float* k1a_b2  = (const float*)d_in[6];
    const float* k2_u    = (const float*)d_in[7];
    const float* k2_v    = (const float*)d_in[8];
    const float* k2_dlog = (const float*)d_in[9];
    const float* k2_alog = (const float*)d_in[10];
    const float* k2_glog = (const float*)d_in[11];
    const float* k2_kern = (const float*)d_in[12];
    const float* k2_pw   = (const float*)d_in[13];
    const float* k2_pb   = (const float*)d_in[14];
    const float* k2_n1w  = (const float*)d_in[15];
    const float* k2_n2w  = (const float*)d_in[16];
    const float* k2_w1   = (const float*)d_in[17];
    const float* k2_b1   = (const float*)d_in[18];
    const float* k2_w2   = (const float*)d_in[19];
    const float* k2_b2   = (const float*)d_in[20];
    const float* k1b_nw  = (const float*)d_in[21];
    const float* k1b_w1  = (const float*)d_in[22];
    const float* k1b_b1  = (const float*)d_in[23];
    const float* k1b_w2  = (const float*)d_in[24];
    const float* k1b_b2  = (const float*)d_in[25];
    const float* k0_nw   = (const float*)d_in[26];
    const float* head_w  = (const float*)d_in[27];
    const float* head_b  = (const float*)d_in[28];
    float* out = (float*)d_out;

    // ---- workspace carve-up ----
    char* p = (char*)d_ws;
    auto alloc = [&](size_t bytes) -> char* {
        char* r = p; p += (bytes + 255) & ~(size_t)255; return r;
    };
    __bf16* wstage = (__bf16*)alloc((size_t)DDIM * VSZ * 2);      // 65.5 MB (max weight)
    float*  h      = (float*) alloc((size_t)TTOK * DDIM * 4);
    float*  hn     = (float*) alloc((size_t)TTOK * DDIM * 4);
    float*  mid    = (float*) alloc((size_t)TTOK * HDIM * 4);
    float*  att    = (float*) alloc((size_t)TTOK * DDIM * 4);
    __bf16* hnT    = (__bf16*)alloc((size_t)BB * DDIM * NTOK * 2);
    float*  sc     = (float*) alloc((size_t)BB * NTOK * NTOK * 4);
    float*  qb     = (float*) alloc((size_t)TTOK * RRK * 4);
    float*  kb     = (float*) alloc((size_t)TTOK * RRK * 4);
    float*  decay  = (float*) alloc((size_t)RRK * NTOK * 4);
    float*  alpha  = (float*) alloc(256);
    float*  gate   = (float*) alloc(256);
    (void)ws_size; (void)in_sizes; (void)n_in; (void)out_size;

    // 1) embedding
    gather_kernel<<<(TTOK * DDIM) / 256, 256, 0, stream>>>(tokens, emb, h);

    // 2) k1a block
    run_k1(h, hn, mid, wstage, k1a_nw, k1a_w1, k1a_b1, k1a_w2, k1a_b2, stream);

    // 3) k2 layers
    for (int l = 0; l < LLAY; l++) {
        const float* u    = k2_u    + (size_t)l * DDIM * RRK;
        const float* v    = k2_v    + (size_t)l * DDIM * RRK;
        const float* dlog = k2_dlog + (size_t)l * RRK;
        const float* alog = k2_alog + (size_t)l * RRK;
        const float* kern = k2_kern + (size_t)l * KKS;
        const float* pw   = k2_pw   + (size_t)l * DDIM * DDIM;
        const float* pb   = k2_pb   + (size_t)l * DDIM;
        const float* n1w  = k2_n1w  + (size_t)l * DDIM;
        const float* n2w  = k2_n2w  + (size_t)l * DDIM;
        const float* w1   = k2_w1   + (size_t)l * DDIM * HDIM;
        const float* b1   = k2_b1   + (size_t)l * HDIM;
        const float* w2   = k2_w2   + (size_t)l * HDIM * DDIM;
        const float* b2   = k2_b2   + (size_t)l * DDIM;

        rmsnorm_kernel<<<TTOK, 256, 0, stream>>>(h, n1w, hn);
        decay_params_kernel<<<(RRK * NTOK) / 256, 256, 0, stream>>>(
            dlog, alog, k2_glog, l, alpha, gate, decay);
        qk_kernel<<<TTOK, 256, 0, stream>>>(hn, u, v, alpha, qb, kb);
        conv_gate_kernel<<<(TTOK * DDIM) / 256, 256, 0, stream>>>(hn, kern, gate, att);
        scores_kernel<<<dim3(NTOK / 16, NTOK / 16, BB), 256, 0, stream>>>(qb, kb, decay, sc);

        // att += scores @ hn   (per batch, WMMA GEMM, K = NTOK)
        for (int b = 0; b < BB; b++) {
            launch_transpose(hn + (size_t)b * NTOK * DDIM, hnT + (size_t)b * DDIM * NTOK,
                             NTOK, DDIM, stream);
            launch_gemm(sc + (size_t)b * NTOK * NTOK, hnT + (size_t)b * DDIM * NTOK,
                        nullptr, att + (size_t)b * NTOK * DDIM, att + (size_t)b * NTOK * DDIM,
                        NTOK, DDIM, NTOK, 0, stream);
        }

        // h = h + (att @ pw + pb)
        launch_transpose(pw, wstage, DDIM, DDIM, stream);
        launch_gemm(att, wstage, pb, h, h, TTOK, DDIM, DDIM, 0, stream);

        // h = h + mlp(rmsnorm(h, n2w))
        run_k1(h, hn, mid, wstage, n2w, w1, b1, w2, b2, stream);
    }

    // 4) k1b block
    run_k1(h, hn, mid, wstage, k1b_nw, k1b_w1, k1b_b1, k1b_w2, k1b_b2, stream);

    // 5) head: out = rmsnorm(h, k0_nw) @ head_w + head_b
    rmsnorm_kernel<<<TTOK, 256, 0, stream>>>(h, k0_nw, hn);
    launch_transpose(head_w, wstage, DDIM, VSZ, stream);
    launch_gemm(hn, wstage, head_b, nullptr, out, TTOK, VSZ, DDIM, 0, stream);
}
